// DGCNN_90297392431377
// MI455X (gfx1250) — matrix operand order repaired and checked by reading, not compiled, not run on hardware
//
#include <hip/hip_runtime.h>
#include <hip/hip_fp16.h>
#include <hip/hip_bf16.h>

typedef __attribute__((ext_vector_type(16))) _Float16 v16h;
typedef __attribute__((ext_vector_type(8)))  _Float16 v8h;
typedef __attribute__((ext_vector_type(8)))  float    v8f;

#define TB 8
#define TN 1024
#define TK 9
#define TT 8192          // TB*TN
#define TE 73728         // TT*TK

// ---------------------------------------------------------------------------
// util
// ---------------------------------------------------------------------------
__global__ void dg_zero_kernel(float* __restrict__ p, int n) {
  int t = blockIdx.x * blockDim.x + threadIdx.x;
  if (t < n) p[t] = 0.f;
}

// ---------------------------------------------------------------------------
// kNN within each graph (1024 candidates). One wave32 per node.
// dist = sq_i + sq_j - 2*dot  (matches reference formula).
// 9 rounds of lexicographic (dist, idx) argmin -> stable top_k tie-break.
// ---------------------------------------------------------------------------
__global__ __launch_bounds__(256) void dg_knn_local(const float* __restrict__ x,
                                                    int* __restrict__ idxout) {
  int gw   = (blockIdx.x * blockDim.x + threadIdx.x) >> 5;
  int lane = threadIdx.x & 31;
  if (gw >= TT) return;
  int i    = gw;
  int base = (i >> 10) << 10;
  float xi0 = x[i*4+0], xi1 = x[i*4+1], xi2 = x[i*4+2], xi3 = x[i*4+3];
  float sqi = xi0*xi0 + xi1*xi1 + xi2*xi2 + xi3*xi3;
  float ds[32];
#pragma unroll
  for (int s = 0; s < 32; ++s) {
    int j = base + s*32 + lane;
    float a = x[j*4+0], b = x[j*4+1], c = x[j*4+2], d = x[j*4+3];
    float sqj = a*a + b*b + c*c + d*d;
    ds[s] = sqi + sqj - 2.f*(xi0*a + xi1*b + xi2*c + xi3*d);
  }
  for (int r = 0; r < TK; ++r) {
    float bm = 3.0e38f; int bs = 0;
#pragma unroll
    for (int s = 0; s < 32; ++s)
      if (ds[s] < bm) { bm = ds[s]; bs = s; }
    int bj = base + bs*32 + lane;
#pragma unroll
    for (int off = 16; off >= 1; off >>= 1) {
      float om = __shfl_xor(bm, off, 32);
      int   oj = __shfl_xor(bj, off, 32);
      if (om < bm || (om == bm && oj < bj)) { bm = om; bj = oj; }
    }
    int wl   = (bj - base) & 31;
    int wsid = (bj - base) >> 5;
    if (lane == wl) {
#pragma unroll
      for (int s = 0; s < 32; ++s)
        if (s == wsid) ds[s] = 3.0e38f;
    }
    if (lane == 0) idxout[i*TK + r] = bj;
  }
}

// ---------------------------------------------------------------------------
// global kNN over all 8192 nodes. One wave32 per node.
// Streamed per-lane sorted top-9, then cross-lane merge (lexicographic).
// ---------------------------------------------------------------------------
__global__ __launch_bounds__(256) void dg_knn_global(const float* __restrict__ x,
                                                     int* __restrict__ idxout) {
  int gw   = (blockIdx.x * blockDim.x + threadIdx.x) >> 5;
  int lane = threadIdx.x & 31;
  if (gw >= TT) return;
  int i = gw;
  float xi0 = x[i*4+0], xi1 = x[i*4+1], xi2 = x[i*4+2], xi3 = x[i*4+3];
  float sqi = xi0*xi0 + xi1*xi1 + xi2*xi2 + xi3*xi3;
  float bd[TK]; int bj[TK];
#pragma unroll
  for (int q = 0; q < TK; ++q) { bd[q] = 3.0e38f; bj[q] = 0x7fffffff; }
  for (int s = 0; s < TT/32; ++s) {
    int j = s*32 + lane;
    float a = x[j*4+0], b = x[j*4+1], c = x[j*4+2], d = x[j*4+3];
    float sqj = a*a + b*b + c*c + d*d;
    float dist = sqi + sqj - 2.f*(xi0*a + xi1*b + xi2*c + xi3*d);
    if (dist < bd[TK-1]) {
      bd[TK-1] = dist; bj[TK-1] = j;
#pragma unroll
      for (int q = TK-1; q > 0; --q) {
        if (bd[q] < bd[q-1]) {  // strict: ties keep earlier (smaller) index first
          float td = bd[q]; bd[q] = bd[q-1]; bd[q-1] = td;
          int   tj = bj[q]; bj[q] = bj[q-1]; bj[q-1] = tj;
        }
      }
    }
  }
  int p = 0;
  for (int r = 0; r < TK; ++r) {
    float cm = 3.0e38f; int cj = 0x7fffffff;
#pragma unroll
    for (int q = 0; q < TK; ++q)
      if (q == p) { cm = bd[q]; cj = bj[q]; }
    float m = cm; int mj = cj;
#pragma unroll
    for (int off = 16; off >= 1; off >>= 1) {
      float om = __shfl_xor(m, off, 32);
      int   oj = __shfl_xor(mj, off, 32);
      if (om < m || (om == m && oj < mj)) { m = om; mj = oj; }
    }
    if (p < TK && cj == mj) ++p;     // j is unique -> unique winner
    if (lane == 0) idxout[i*TK + r] = mj;
  }
}

// ---------------------------------------------------------------------------
// Edge-conv GEMM layer using v_wmma_f32_16x16x32_f16.
//   EDGE_IN : A rows are [x_i, x_j - x_i] (8 dims, zero-padded to K=32)
//   else    : A rows are relu(Hin * mul + add) from f16 hidden buffer
// Weights are staged in LDS *pre-swizzled into the WMMA B-fragment layout*
// (lane-major, 16 contiguous halves per lane, 32B aligned) so each fragment
// is 2x ds_load_b128 instead of 16x ds_load_u16.
// Output: Hout (f16, [E, COUT]) and optional per-channel sum/sumsq (BN stats).
// Block: 128 threads = 4 waves. COUT>=64: block = 16 rows, waves split columns.
// COUT==16: block = 64 rows, one row-tile per wave.
// ---------------------------------------------------------------------------
template<int CIN, int COUT, bool EDGE_IN, bool STATS>
__global__ __launch_bounds__(128) void dg_edge_gemm(
    const float* __restrict__ xin, const int* __restrict__ idx,
    const _Float16* __restrict__ Hin,
    const float* __restrict__ mulv, const float* __restrict__ addv,
    const float* __restrict__ W,   const float* __restrict__ bias,
    _Float16* __restrict__ Hout,
    float* __restrict__ stSum, float* __restrict__ stSqs) {
  constexpr int CINP = EDGE_IN ? 32 : CIN;             // K padded to mult of 32
  constexpr int RT   = (COUT == 16) ? 4 : 1;           // row-tiles per block
  constexpr int TPW  = (COUT >= 64) ? (COUT / 64) : 1; // col-tiles per wave
  constexpr int NKC  = CINP / 32;                      // K chunks
  constexpr int NTL  = COUT / 16;                      // column tiles total
  __shared__ __align__(32) _Float16 sA[RT * 16 * CINP];
  __shared__ __align__(32) _Float16 sWf[NTL * NKC * 32 * 16]; // B fragments
  const int tid = threadIdx.x;

  // Stage weights directly in WMMA B-fragment order:
  //   fragment f = tile*NKC + kc ; lane ln ; element e (0..15)
  //   k = kc*32 + (e & 7) + (e >= 8 ? 16 : 0) + (ln>=16 ? 8 : 0), n = tile*16 + (ln & 15)
  for (int t = tid; t < NTL * NKC * 512; t += 128) {
    int e    = t & 15;
    int ln   = (t >> 4) & 31;
    int f    = t >> 9;
    int kc   = f % NKC;
    int tile = f / NKC;
    int k = kc * 32 + (e & 7) + ((e >= 8) ? 16 : 0) + ((ln >= 16) ? 8 : 0);
    int n = tile * 16 + (ln & 15);
    float wv;
    if (EDGE_IN) wv = (k < 8) ? W[k * COUT + n] : 0.f;
    else         wv = W[k * COUT + n];
    sWf[t] = (_Float16)wv;
  }
  // Stage A rows (row-major, CINP leading dim)
  for (int t = tid; t < RT * 16 * CINP; t += 128) {
    int rr = t / CINP;
    int c  = t - rr * CINP;
    long e = (long)blockIdx.x * (RT * 16) + rr;
    float v;
    if (EDGE_IN) {
      v = 0.f;
      if (c < 8) {
        int ii = (int)(e / TK);
        int jj = idx[e];
        int d  = c & 3;
        float xiv = xin[ii * 4 + d];
        v = (c < 4) ? xiv : (xin[jj * 4 + d] - xiv);
      }
    } else {
      float h = (float)Hin[e * CIN + c];
      h = h * mulv[c] + addv[c];
      v = fmaxf(h, 0.f);
    }
    sA[t] = (_Float16)v;
  }
  __syncthreads();

  const int lane = tid & 31;
  const int wv   = tid >> 5;
  const int rt   = (COUT == 16) ? wv : 0;
  const int m    = lane & 15;
  const int hi   = lane >> 4;
#pragma unroll
  for (int tt = 0; tt < TPW; ++tt) {
    const int tileIdx = (COUT == 16) ? 0 : (wv * TPW + tt);
    const int col0    = tileIdx * 16;
    const int n       = col0 + m;
    v8f acc;
    float bc = bias[n];
#pragma unroll
    for (int r = 0; r < 8; ++r) acc[r] = bc;
#pragma unroll
    for (int kc = 0; kc < NKC; ++kc) {
      // A fragment: 2x 16B vector loads (ISA 16-bit A 16x32 layout)
      const _Float16* rowp = &sA[rt * 16 * CINP + m * CINP + kc * 32 + hi * 8];
      v8h alo = *(const v8h*)rowp;
      v8h ahi = *(const v8h*)(rowp + 16);
      v16h a;
#pragma unroll
      for (int e2 = 0; e2 < 8; ++e2) { a[e2] = alo[e2]; a[e2 + 8] = ahi[e2]; }
      // B fragment: one 32B contiguous per-lane load (pre-swizzled)
      v16h bmat = *(const v16h*)&sWf[((tileIdx * NKC + kc) * 32 + lane) * 16];
      acc = __builtin_amdgcn_wmma_f32_16x16x32_f16(false, a, false, bmat,
                                                   (short)0, acc, false, false);
    }
    long e0 = (long)blockIdx.x * (RT * 16) + rt * 16;
    float ps = 0.f, pq = 0.f;
#pragma unroll
    for (int r = 0; r < 8; ++r) {
      float vv = acc[r];                       // row = e0 + r + hi*8, col = n
      Hout[(e0 + r + hi * 8) * COUT + n] = (_Float16)vv;
      ps += vv; pq += vv * vv;
    }
    if (STATS) {
      ps += __shfl_xor(ps, 16, 32);            // lane L and L+16 share column n
      pq += __shfl_xor(pq, 16, 32);
      if (hi == 0) {
        atomicAdd(&stSum[n], ps);
        atomicAdd(&stSqs[n], pq);
      }
    }
  }
}

// per-channel BN fold: mul = g*rsqrt(var+eps), add = beta - mu*mul
__global__ void dg_bn_finalize(const float* __restrict__ sum,
                               const float* __restrict__ sqs,
                               const float* __restrict__ g,
                               const float* __restrict__ be,
                               float* __restrict__ mulv, float* __restrict__ addv,
                               int C, float invN) {
  int c = blockIdx.x * blockDim.x + threadIdx.x;
  if (c < C) {
    float mu  = sum[c] * invN;
    float var = sqs[c] * invN - mu * mu;
    float mm  = g[c] * rsqrtf(fmaxf(var, 0.f) + 1e-5f);
    mulv[c] = mm;
    addv[c] = be[c] - mu * mm;
  }
}

// mean over K=9 edges per node: [E,C] f16 -> [T,C] f32
template<int C>
__global__ void dg_mean_k(const _Float16* __restrict__ H, float* __restrict__ out) {
  int t = blockIdx.x * blockDim.x + threadIdx.x;
  if (t >= TT * C) return;
  int i = t / C, c = t - i * C;
  float s = 0.f;
#pragma unroll
  for (int k = 0; k < TK; ++k) s += (float)H[((long)i * TK + k) * C + c];
  out[t] = s * (1.f / 9.f);
}

// head MLP 128->64->32->1 per node (VALU, LDS-resident weights) + global stats
__global__ __launch_bounds__(256) void dg_head(
    const float* __restrict__ x1,
    const float* __restrict__ W1, const float* __restrict__ b1,
    const float* __restrict__ W2, const float* __restrict__ b2,
    const float* __restrict__ W3, const float* __restrict__ b3,
    float* __restrict__ h3buf, float* __restrict__ gstats) {
  __shared__ float sW1[128 * 64];
  __shared__ float sW2[64 * 32];
  __shared__ float sW3[32];
  __shared__ float sb1[64], sb2[32];
  int tid = threadIdx.x;
  for (int t = tid; t < 128 * 64; t += 256) sW1[t] = W1[t];
  for (int t = tid; t < 64 * 32; t += 256) sW2[t] = W2[t];
  if (tid < 32) sW3[tid] = W3[tid];
  if (tid < 64) sb1[tid] = b1[tid];
  if (tid < 32) sb2[tid] = b2[tid];
  __syncthreads();
  int i = blockIdx.x * 256 + tid;
  float h1[64];
#pragma unroll
  for (int o = 0; o < 64; ++o) h1[o] = sb1[o];
  for (int c = 0; c < 128; ++c) {
    float xv = x1[(long)i * 128 + c];
#pragma unroll
    for (int o = 0; o < 64; ++o) h1[o] += xv * sW1[c * 64 + o];
  }
  float h2[32];
#pragma unroll
  for (int o = 0; o < 32; ++o) h2[o] = sb2[o];
#pragma unroll
  for (int c = 0; c < 64; ++c) {
    float v = fmaxf(h1[c], 0.f);
#pragma unroll
    for (int o = 0; o < 32; ++o) h2[o] += v * sW2[c * 32 + o];
  }
  float h3 = b3[0];
#pragma unroll
  for (int c = 0; c < 32; ++c) h3 += fmaxf(h2[c], 0.f) * sW3[c];
  h3buf[i] = h3;
  atomicAdd(&gstats[0], h3);
  atomicAdd(&gstats[1], h3 * h3);
}

// normalize (mean / unbiased std) -> sigmoid gate -> xl, xs
__global__ void dg_gate(const float* __restrict__ h3buf,
                        const float* __restrict__ gstats,
                        const float* __restrict__ x,
                        float* __restrict__ xl, float* __restrict__ xs) {
  int i = blockIdx.x * blockDim.x + threadIdx.x;
  if (i >= TT) return;
  float s = gstats[0], sq = gstats[1];
  float mean = s / (float)TT;
  float var  = (sq - s * s / (float)TT) / (float)(TT - 1);
  float sd   = sqrtf(fmaxf(var, 0.f));
  float z    = (h3buf[i] - mean) / (sd + 1e-5f);
  float g    = 1.f / (1.f + expf(-z));
#pragma unroll
  for (int d = 0; d < 4; ++d) {
    float xv = x[i * 4 + d];
    xl[i * 4 + d] = g * xv;
    xs[i * 4 + d] = (1.f - g) * xv;
  }
}

// per-graph max pool of featL/featS [T,16] + final 32->1 linear -> out[8]
__global__ __launch_bounds__(256) void dg_pool_final(
    const float* __restrict__ featL, const float* __restrict__ featS,
    const float* __restrict__ w, const float* __restrict__ b,
    float* __restrict__ out) {
  __shared__ float sL[256], sS[256];
  int g = blockIdx.x, tid = threadIdx.x;
  int c = tid & 15, rg = tid >> 4;
  float ml = -3.0e38f, ms = -3.0e38f;
  for (int q = 0; q < 64; ++q) {
    int i = g * 1024 + rg * 64 + q;
    ml = fmaxf(ml, featL[i * 16 + c]);
    ms = fmaxf(ms, featS[i * 16 + c]);
  }
  sL[tid] = ml; sS[tid] = ms;
  __syncthreads();
  for (int st = 128; st >= 16; st >>= 1) {
    if (tid < st) {
      sL[tid] = fmaxf(sL[tid], sL[tid + st]);
      sS[tid] = fmaxf(sS[tid], sS[tid + st]);
    }
    __syncthreads();
  }
  if (tid == 0) {
    float acc = b[0];
    for (int cc = 0; cc < 16; ++cc)
      acc += sL[cc] * w[cc] + sS[cc] * w[16 + cc];
    out[g] = acc;
  }
}

// ---------------------------------------------------------------------------
extern "C" void kernel_launch(void* const* d_in, const int* in_sizes, int n_in,
                              void* d_out, int out_size, void* d_ws, size_t ws_size,
                              hipStream_t stream) {
  (void)in_sizes; (void)n_in; (void)out_size; (void)ws_size;
  const float* x      = (const float*)d_in[0];
  const float* c1_W1  = (const float*)d_in[1];
  const float* c1_b1  = (const float*)d_in[2];
  const float* c1_g1  = (const float*)d_in[3];
  const float* c1_be1 = (const float*)d_in[4];
  const float* c1_W2  = (const float*)d_in[5];
  const float* c1_b2  = (const float*)d_in[6];
  const float* c1_g2  = (const float*)d_in[7];
  const float* c1_be2 = (const float*)d_in[8];
  const float* c1_W3  = (const float*)d_in[9];
  const float* c1_b3  = (const float*)d_in[10];
  const float* c3_W1  = (const float*)d_in[11];
  const float* c3_b1  = (const float*)d_in[12];
  const float* c3_g1  = (const float*)d_in[13];
  const float* c3_be1 = (const float*)d_in[14];
  const float* c3_W2  = (const float*)d_in[15];
  const float* c3_b2  = (const float*)d_in[16];
  const float* c3_g2  = (const float*)d_in[17];
  const float* c3_be2 = (const float*)d_in[18];
  const float* c3_W3  = (const float*)d_in[19];
  const float* c3_b3  = (const float*)d_in[20];
  const float* h_W1   = (const float*)d_in[21];
  const float* h_b1   = (const float*)d_in[22];
  const float* h_W2   = (const float*)d_in[23];
  const float* h_b2   = (const float*)d_in[24];
  const float* h_W3   = (const float*)d_in[25];
  const float* h_b3   = (const float*)d_in[26];
  const float* lin2_W = (const float*)d_in[27];
  const float* lin2_b = (const float*)d_in[28];
  float* out = (float*)d_out;

  // workspace layout
  float* stats  = (float*)d_ws;                 // 2048 f32 (BN sums + head stats)
  float* mulAdd = stats + 2048;                 // 1024 f32 (folded BN scale/shift)
  int*   idx1   = (int*)(mulAdd + 1024);        // [E]
  int*   idxL   = idx1 + TE;                    // [E]
  int*   idxS   = idxL + TE;                    // [E]
  float* h3buf  = (float*)(idxS + TE);          // [T]
  float* x1     = h3buf + TT;                   // [T,128]
  float* xl     = x1 + (size_t)TT * 128;        // [T,4]
  float* xs     = xl + (size_t)TT * 4;          // [T,4]
  float* featL  = xs + (size_t)TT * 4;          // [T,16]
  float* featS  = featL + (size_t)TT * 16;      // [T,16]
  _Float16* HA  = (_Float16*)(featS + (size_t)TT * 16); // [E,128] f16
  _Float16* HB  = HA + (size_t)TE * 128;                 // [E,128] f16

  float* s_c1a = stats + 0;    // sum[128] @0, sqs[128] @128
  float* s_c1b = stats + 256;  // sum @256, sqs @384
  float* s_l1  = stats + 512;  // 64 + 64
  float* s_l2  = stats + 640;
  float* s_s1  = stats + 768;
  float* s_s2  = stats + 896;
  float* s_hd  = stats + 1024; // sum, sumsq of head output
  float* mul1 = mulAdd + 0,   *add1 = mulAdd + 128;
  float* mul2 = mulAdd + 256, *add2 = mulAdd + 384;
  float* mc3a = mulAdd + 512, *ac3a = mulAdd + 576;
  float* mc3b = mulAdd + 640, *ac3b = mulAdd + 704;

  const float invE = 1.f / (float)TE;

  dg_zero_kernel<<<8, 256, 0, stream>>>(stats, 2048);

  // ---- conv1: batch-local kNN + EdgeConv(8->128->128->128), mean over K ----
  dg_knn_local<<<TT/8, 256, 0, stream>>>(x, idx1);

  dg_edge_gemm<8, 128, true, true><<<TE/16, 128, 0, stream>>>(
      x, idx1, nullptr, nullptr, nullptr, c1_W1, c1_b1, HA, s_c1a, s_c1a + 128);
  dg_bn_finalize<<<1, 128, 0, stream>>>(s_c1a, s_c1a + 128, c1_g1, c1_be1,
                                        mul1, add1, 128, invE);
  dg_edge_gemm<128, 128, false, true><<<TE/16, 128, 0, stream>>>(
      nullptr, nullptr, HA, mul1, add1, c1_W2, c1_b2, HB, s_c1b, s_c1b + 128);
  dg_bn_finalize<<<1, 128, 0, stream>>>(s_c1b, s_c1b + 128, c1_g2, c1_be2,
                                        mul2, add2, 128, invE);
  dg_edge_gemm<128, 128, false, false><<<TE/16, 128, 0, stream>>>(
      nullptr, nullptr, HB, mul2, add2, c1_W3, c1_b3, HA, nullptr, nullptr);
  dg_mean_k<128><<<(TT*128)/256, 256, 0, stream>>>(HA, x1);

  // ---- head MLP + global normalize + sigmoid gate ----
  dg_head<<<TT/256, 256, 0, stream>>>(x1, h_W1, h_b1, h_W2, h_b2, h_W3, h_b3,
                                      h3buf, s_hd);
  dg_gate<<<TT/256, 256, 0, stream>>>(h3buf, s_hd, x, xl, xs);

  // ---- conv3 on xl (global kNN, 8->64->64->16), mean, feats ----
  dg_knn_global<<<TT/8, 256, 0, stream>>>(xl, idxL);
  dg_edge_gemm<8, 64, true, true><<<TE/16, 128, 0, stream>>>(
      xl, idxL, nullptr, nullptr, nullptr, c3_W1, c3_b1, HA, s_l1, s_l1 + 64);
  dg_bn_finalize<<<1, 128, 0, stream>>>(s_l1, s_l1 + 64, c3_g1, c3_be1,
                                        mc3a, ac3a, 64, invE);
  dg_edge_gemm<64, 64, false, true><<<TE/16, 128, 0, stream>>>(
      nullptr, nullptr, HA, mc3a, ac3a, c3_W2, c3_b2, HB, s_l2, s_l2 + 64);
  dg_bn_finalize<<<1, 128, 0, stream>>>(s_l2, s_l2 + 64, c3_g2, c3_be2,
                                        mc3b, ac3b, 64, invE);
  dg_edge_gemm<64, 16, false, false><<<TE/64, 128, 0, stream>>>(
      nullptr, nullptr, HB, mc3b, ac3b, c3_W3, c3_b3, HA, nullptr, nullptr);
  dg_mean_k<16><<<(TT*16)/256, 256, 0, stream>>>(HA, featL);

  // ---- conv3 on xs ----
  dg_knn_global<<<TT/8, 256, 0, stream>>>(xs, idxS);
  dg_edge_gemm<8, 64, true, true><<<TE/16, 128, 0, stream>>>(
      xs, idxS, nullptr, nullptr, nullptr, c3_W1, c3_b1, HA, s_s1, s_s1 + 64);
  dg_bn_finalize<<<1, 128, 0, stream>>>(s_s1, s_s1 + 64, c3_g1, c3_be1,
                                        mc3a, ac3a, 64, invE);
  dg_edge_gemm<64, 64, false, true><<<TE/16, 128, 0, stream>>>(
      nullptr, nullptr, HA, mc3a, ac3a, c3_W2, c3_b2, HB, s_s2, s_s2 + 64);
  dg_bn_finalize<<<1, 128, 0, stream>>>(s_s2, s_s2 + 64, c3_g2, c3_be2,
                                        mc3b, ac3b, 64, invE);
  dg_edge_gemm<64, 16, false, false><<<TE/64, 128, 0, stream>>>(
      nullptr, nullptr, HB, mc3b, ac3b, c3_W3, c3_b3, HA, nullptr, nullptr);
  dg_mean_k<16><<<(TT*16)/256, 256, 0, stream>>>(HA, featS);

  // ---- per-graph max pool + final linear ----
  dg_pool_final<<<TB, 256, 0, stream>>>(featL, featS, lin2_W, lin2_b, out);
}
